// decoderSelfAttention_52072183497347
// MI455X (gfx1250) — matrix operand order repaired and checked by reading, not compiled, not run on hardware
//
#include <hip/hip_runtime.h>

// ---------------- types for WMMA ----------------
typedef __bf16 bf16_t;
typedef bf16_t v16bf __attribute__((ext_vector_type(16)));
typedef float  v8f   __attribute__((ext_vector_type(8)));

union Frag16 { v16bf v; uint4 q[2]; };
union Pack8  { unsigned short u[8]; uint4 q; };

__device__ __forceinline__ unsigned short f2bf(float f) {
  union { float f; unsigned u; } c; c.f = f;
  unsigned r = c.u + 0x7FFFu + ((c.u >> 16) & 1u);   // round-to-nearest-even
  return (unsigned short)(r >> 16);
}
__device__ __forceinline__ float bf2f(unsigned short h) {
  union { unsigned u; float f; } c; c.u = ((unsigned)h) << 16; return c.f;
}

// ----- CDNA5 async global->LDS copy (ASYNCcnt path), inline asm -----
__device__ __forceinline__ void async_copy_b128(void* lds, const void* g) {
#if defined(__AMDGCN__)
  asm volatile("global_load_async_to_lds_b128 %0, %1, off"
               :: "v"((unsigned)(unsigned long long)lds), "v"(g)
               : "memory");
#endif
}
__device__ __forceinline__ void wait_async0() {
#if defined(__AMDGCN__)
  asm volatile("s_wait_asynccnt 0" ::: "memory");
#endif
}

#define BATCH 8
#define SEQ   2048
#define HID   1024
#define MTOT  (BATCH * SEQ)      // 16384

#define TM 128
#define TN 64
#define TK 32
#define LDA 40                    // padded LDS stride (bf16 elems)
#define LDB 40

// ---------------- 1a: fp32 -> bf16 convert ----------------
__global__ void k_convert(const float* __restrict__ src,
                          unsigned short* __restrict__ dst, int n) {
  int i = (blockIdx.x * blockDim.x + threadIdx.x) * 4;
  if (i + 3 < n) {
    float4 f = *(const float4*)(src + i);
    ushort4 o;
    o.x = f2bf(f.x); o.y = f2bf(f.y); o.z = f2bf(f.z); o.w = f2bf(f.w);
    *(ushort4*)(dst + i) = o;
  }
}

// ---------------- 1b: transpose + convert weights (H x H) ----------------
__global__ void k_transpose_convert(const float* __restrict__ w,
                                    unsigned short* __restrict__ wT) {
  __shared__ float tile[32][33];
  int bx = blockIdx.x * 32;   // n (column of w)
  int by = blockIdx.y * 32;   // k (row of w)
  int tx = threadIdx.x, ty = threadIdx.y;   // 32 x 8
#pragma unroll
  for (int j = 0; j < 32; j += 8)
    tile[ty + j][tx] = w[(size_t)(by + ty + j) * HID + bx + tx];
  __syncthreads();
#pragma unroll
  for (int j = 0; j < 32; j += 8)
    wT[(size_t)(bx + ty + j) * HID + by + tx] = f2bf(tile[tx][ty + j]);
}

// ---- fragment helpers (ISA VGPR layouts for 16-bit A / B operands) ----
__device__ __forceinline__ void load_a_frag(Frag16& a, const unsigned short* base,
                                            int row, int lane) {
  int ak = (lane < 16) ? 0 : 8;
  const unsigned short* p = base + row * LDA + ak;
  a.q[0] = *(const uint4*)(p);
  a.q[1] = *(const uint4*)(p + 16);
}
__device__ __forceinline__ void load_b_frag(Frag16& b, const unsigned short* base,
                                            int col, int lane) {
  int bk = (lane < 16) ? 0 : 8;
  const unsigned short* p = base + col * LDB + bk;
  b.q[0] = *(const uint4*)(p);
  b.q[1] = *(const uint4*)(p + 16);
}

// ---------------- 2a: fused Q+K GEMM (bf16 WMMA, fp32 acc) ----------------
// Qb, Kb row-major [MTOT, H]
__global__ __launch_bounds__(256) void k_qk_gemm(
    const unsigned short* __restrict__ xb,
    const unsigned short* __restrict__ wqT, const unsigned short* __restrict__ wkT,
    const float* __restrict__ bq, const float* __restrict__ bk,
    unsigned short* __restrict__ Qb, unsigned short* __restrict__ Kb) {
  __shared__ unsigned short xs[2][TM * LDA];
  __shared__ unsigned short wsm[2][2][TN * LDB];

  int m0 = blockIdx.x * TM;
  int n0 = blockIdx.y * TN;
  int tid = threadIdx.x;
  int lane = tid & 31, wave = tid >> 5;
  int wm = (wave & 3) * 32, wn = (wave >> 2) * 32;

  const unsigned short* xg  = xb  + (size_t)m0 * HID;
  const unsigned short* wg0 = wqT + (size_t)n0 * HID;
  const unsigned short* wg1 = wkT + (size_t)n0 * HID;

  auto stage = [&](int buf, int k0) {
#pragma unroll
    for (int i = 0; i < 2; ++i) {               // x tile: 128 rows x 32 cols
      int idx = tid + i * 256;
      int r = idx >> 2, c = (idx & 3) * 8;
      async_copy_b128(&xs[buf][r * LDA + c], &xg[(size_t)r * HID + k0 + c]);
    }
    int r = tid >> 2, c = (tid & 3) * 8;        // 2 weight tiles: 64 x 32 each
    async_copy_b128(&wsm[buf][0][r * LDB + c], &wg0[(size_t)r * HID + k0 + c]);
    async_copy_b128(&wsm[buf][1][r * LDB + c], &wg1[(size_t)r * HID + k0 + c]);
  };

  v8f acc[2][2][2] = {};

  auto compute = [&](const unsigned short* xsb, const unsigned short* ws0,
                     const unsigned short* ws1) {
    Frag16 a[2];
    int am = wm + (lane & 15);
#pragma unroll
    for (int ti = 0; ti < 2; ++ti)
      load_a_frag(a[ti], xsb, am + ti * 16, lane);
    int bn = wn + (lane & 15);
#pragma unroll
    for (int wi = 0; wi < 2; ++wi) {
      const unsigned short* wsb = (wi == 0) ? ws0 : ws1;
#pragma unroll
      for (int tj = 0; tj < 2; ++tj) {
        Frag16 bf;
        load_b_frag(bf, wsb, bn + tj * 16, lane);
#pragma unroll
        for (int ti = 0; ti < 2; ++ti)
          acc[wi][ti][tj] = __builtin_amdgcn_wmma_f32_16x16x32_bf16(
              false, a[ti].v, false, bf.v, (short)0, acc[wi][ti][tj],
              false, false);
      }
    }
  };

  const int NIT = HID / TK;                      // 32 (even)
  stage(0, 0);
  for (int it = 0; it < NIT - 2; it += 2) {
    wait_async0(); __syncthreads();
    stage(1, (it + 1) * TK);
    compute(xs[0], wsm[0][0], wsm[0][1]);
    wait_async0(); __syncthreads();
    stage(0, (it + 2) * TK);
    compute(xs[1], wsm[1][0], wsm[1][1]);
  }
  wait_async0(); __syncthreads();
  stage(1, (NIT - 1) * TK);
  compute(xs[0], wsm[0][0], wsm[0][1]);
  wait_async0(); __syncthreads();
  compute(xs[1], wsm[1][0], wsm[1][1]);

#pragma unroll
  for (int wi = 0; wi < 2; ++wi) {
#pragma unroll
    for (int tj = 0; tj < 2; ++tj) {
      int n = n0 + wn + tj * 16 + (lane & 15);
      float bias = (wi == 0) ? bq[n] : bk[n];
      unsigned short* outp = (wi == 0) ? Qb : Kb;
#pragma unroll
      for (int ti = 0; ti < 2; ++ti) {
        v8f c = acc[wi][ti][tj];
        int mbase = m0 + wm + ti * 16 + ((lane < 16) ? 0 : 8);
#pragma unroll
        for (int r = 0; r < 8; ++r)
          outp[(size_t)(mbase + r) * HID + n] = f2bf(c[r] + bias);
      }
    }
  }
}

// ---------------- 2b: V GEMM, stored transposed Vt [B][H][S] ----------------
__global__ __launch_bounds__(256) void k_v_gemm(
    const unsigned short* __restrict__ xb,
    const unsigned short* __restrict__ wvT, const float* __restrict__ bv,
    unsigned short* __restrict__ Vt) {
  __shared__ unsigned short xs[2][TM * LDA];
  __shared__ unsigned short wsm[2][TN * LDB];

  int m0 = blockIdx.x * TM;
  int n0 = blockIdx.y * TN;
  int tid = threadIdx.x;
  int lane = tid & 31, wave = tid >> 5;
  int wm = (wave & 3) * 32, wn = (wave >> 2) * 32;

  const unsigned short* xg = xb  + (size_t)m0 * HID;
  const unsigned short* wg = wvT + (size_t)n0 * HID;

  auto stage = [&](int buf, int k0) {
#pragma unroll
    for (int i = 0; i < 2; ++i) {
      int idx = tid + i * 256;
      int r = idx >> 2, c = (idx & 3) * 8;
      async_copy_b128(&xs[buf][r * LDA + c], &xg[(size_t)r * HID + k0 + c]);
    }
    int r = tid >> 2, c = (tid & 3) * 8;
    async_copy_b128(&wsm[buf][r * LDB + c], &wg[(size_t)r * HID + k0 + c]);
  };

  v8f acc[2][2] = {};

  auto compute = [&](const unsigned short* xsb, const unsigned short* wsb) {
    Frag16 a[2], bf[2];
    int am = wm + (lane & 15);
    int bn = wn + (lane & 15);
#pragma unroll
    for (int ti = 0; ti < 2; ++ti) {
      load_a_frag(a[ti], xsb, am + ti * 16, lane);
      load_b_frag(bf[ti], wsb, bn + ti * 16, lane);
    }
#pragma unroll
    for (int ti = 0; ti < 2; ++ti)
#pragma unroll
      for (int tj = 0; tj < 2; ++tj)
        acc[ti][tj] = __builtin_amdgcn_wmma_f32_16x16x32_bf16(
            false, a[ti].v, false, bf[tj].v, (short)0, acc[ti][tj],
            false, false);
  };

  const int NIT = HID / TK;
  stage(0, 0);
  for (int it = 0; it < NIT - 2; it += 2) {
    wait_async0(); __syncthreads();
    stage(1, (it + 1) * TK);
    compute(xs[0], wsm[0]);
    wait_async0(); __syncthreads();
    stage(0, (it + 2) * TK);
    compute(xs[1], wsm[1]);
  }
  wait_async0(); __syncthreads();
  stage(1, (NIT - 1) * TK);
  compute(xs[0], wsm[0]);
  wait_async0(); __syncthreads();
  compute(xs[1], wsm[1]);

#pragma unroll
  for (int tj = 0; tj < 2; ++tj) {
    int n = n0 + wn + tj * 16 + (lane & 15);
    float bias = bv[n];
#pragma unroll
    for (int ti = 0; ti < 2; ++ti) {
      v8f c = acc[ti][tj];
      int mbase = m0 + wm + ti * 16 + ((lane < 16) ? 0 : 8);
      int b = mbase / SEQ, s = mbase % SEQ;   // tile lies inside one batch
      Pack8 pk;
#pragma unroll
      for (int r = 0; r < 8; ++r) pk.u[r] = f2bf(c[r] + bias);
      *(uint4*)(&Vt[((size_t)b * HID + n) * SEQ + s]) = pk.q;
    }
  }
}

// ---------------- 3: scores = scale * Q K^T + mask, bf16 ----------------
__global__ __launch_bounds__(256) void k_scores(
    const unsigned short* __restrict__ Qb, const unsigned short* __restrict__ Kb,
    const int* __restrict__ mask, unsigned short* __restrict__ P) {
  __shared__ unsigned short qs[2][TM * LDA];
  __shared__ unsigned short ks[2][TN * LDB];
  __shared__ int mrow[TM];

  int b = blockIdx.z;
  int s0 = blockIdx.x * TM, t0 = blockIdx.y * TN;
  int tid = threadIdx.x;
  int lane = tid & 31, wave = tid >> 5;
  int wm = (wave & 3) * 32, wn = (wave >> 2) * 32;

  if (tid < TM) mrow[tid] = mask[b * SEQ + s0 + tid];

  const unsigned short* qg = Qb + ((size_t)b * SEQ + s0) * HID;
  const unsigned short* kg = Kb + ((size_t)b * SEQ + t0) * HID;

  auto stage = [&](int buf, int k0) {
#pragma unroll
    for (int i = 0; i < 2; ++i) {
      int idx = tid + i * 256;
      int r = idx >> 2, c = (idx & 3) * 8;
      async_copy_b128(&qs[buf][r * LDA + c], &qg[(size_t)r * HID + k0 + c]);
    }
    int r = tid >> 2, c = (tid & 3) * 8;
    async_copy_b128(&ks[buf][r * LDB + c], &kg[(size_t)r * HID + k0 + c]);
  };

  v8f acc[2][2] = {};

  auto compute = [&](const unsigned short* qsb, const unsigned short* ksb) {
    Frag16 a[2], bf[2];
    int am = wm + (lane & 15);
    int bn = wn + (lane & 15);
#pragma unroll
    for (int ti = 0; ti < 2; ++ti) {
      load_a_frag(a[ti], qsb, am + ti * 16, lane);
      load_b_frag(bf[ti], ksb, bn + ti * 16, lane);
    }
#pragma unroll
    for (int ti = 0; ti < 2; ++ti)
#pragma unroll
      for (int tj = 0; tj < 2; ++tj)
        acc[ti][tj] = __builtin_amdgcn_wmma_f32_16x16x32_bf16(
            false, a[ti].v, false, bf[tj].v, (short)0, acc[ti][tj],
            false, false);
  };

  const int NIT = HID / TK;
  stage(0, 0);
  for (int it = 0; it < NIT - 2; it += 2) {
    wait_async0(); __syncthreads();
    stage(1, (it + 1) * TK);
    compute(qs[0], ks[0]);
    wait_async0(); __syncthreads();
    stage(0, (it + 2) * TK);
    compute(qs[1], ks[1]);
  }
  wait_async0(); __syncthreads();
  stage(1, (NIT - 1) * TK);
  compute(qs[0], ks[0]);
  wait_async0(); __syncthreads();
  compute(qs[1], ks[1]);

  const float scale = 0.03125f;  // 1/sqrt(1024)
#pragma unroll
  for (int ti = 0; ti < 2; ++ti)
#pragma unroll
    for (int tj = 0; tj < 2; ++tj) {
      int t = t0 + wn + tj * 16 + (lane & 15);
#pragma unroll
      for (int r = 0; r < 8; ++r) {
        int s_loc = wm + ti * 16 + ((lane < 16) ? 0 : 8) + r;
        int s = s0 + s_loc;
        float add = (t <= s) ? -1e9f : (mrow[s_loc] ? 0.0f : -1e9f);
        P[((size_t)b * SEQ + s) * SEQ + t] = f2bf(acc[ti][tj][r] * scale + add);
      }
    }
}

// ---------------- 4: row softmax, in place, bf16 ----------------
__global__ __launch_bounds__(256) void k_softmax(unsigned short* __restrict__ P) {
  __shared__ float redm[8];
  __shared__ float redsum[8];
  size_t row = blockIdx.x;
  unsigned short* p = P + row * SEQ;
  int tid = threadIdx.x;
  int lane = tid & 31, wave = tid >> 5;

  Pack8 v;
  v.q = *(const uint4*)(p + tid * 8);
  float f[8];
  float mx = -3.4e38f;
#pragma unroll
  for (int i = 0; i < 8; ++i) { f[i] = bf2f(v.u[i]); mx = fmaxf(mx, f[i]); }
#pragma unroll
  for (int o = 16; o > 0; o >>= 1) mx = fmaxf(mx, __shfl_xor(mx, o, 32));
  if (lane == 0) redm[wave] = mx;
  __syncthreads();
  if (wave == 0) {
    float m2 = redm[lane & 7];
#pragma unroll
    for (int o = 4; o > 0; o >>= 1) m2 = fmaxf(m2, __shfl_xor(m2, o, 32));
    if (lane == 0) redm[0] = m2;
  }
  __syncthreads();
  mx = redm[0];

  float sum = 0.0f;
#pragma unroll
  for (int i = 0; i < 8; ++i) { f[i] = __expf(f[i] - mx); sum += f[i]; }
#pragma unroll
  for (int o = 16; o > 0; o >>= 1) sum += __shfl_xor(sum, o, 32);
  if (lane == 0) redsum[wave] = sum;
  __syncthreads();
  if (wave == 0) {
    float s2 = redsum[lane & 7];
#pragma unroll
    for (int o = 4; o > 0; o >>= 1) s2 += __shfl_xor(s2, o, 32);
    if (lane == 0) redsum[0] = s2;
  }
  __syncthreads();
  float inv = 1.0f / redsum[0];
#pragma unroll
  for (int i = 0; i < 8; ++i) v.u[i] = f2bf(f[i] * inv);
  *(uint4*)(p + tid * 8) = v.q;
}

// ---------------- 5: O = P @ V (Vt is [B][H][S]) ----------------
__global__ __launch_bounds__(256) void k_out_gemm(
    const unsigned short* __restrict__ P, const unsigned short* __restrict__ Vt,
    float* __restrict__ O) {
  __shared__ unsigned short ps[2][TM * LDA];
  __shared__ unsigned short vs[2][TN * LDB];

  int b = blockIdx.z;
  int s0 = blockIdx.x * TM, h0 = blockIdx.y * TN;
  int tid = threadIdx.x;
  int lane = tid & 31, wave = tid >> 5;
  int wm = (wave & 3) * 32, wn = (wave >> 2) * 32;

  const unsigned short* pg = P  + ((size_t)b * SEQ + s0) * SEQ;
  const unsigned short* vg = Vt + ((size_t)b * HID + h0) * SEQ;

  auto stage = [&](int buf, int t0k) {
#pragma unroll
    for (int i = 0; i < 2; ++i) {
      int idx = tid + i * 256;
      int r = idx >> 2, c = (idx & 3) * 8;
      async_copy_b128(&ps[buf][r * LDA + c], &pg[(size_t)r * SEQ + t0k + c]);
    }
    int r = tid >> 2, c = (tid & 3) * 8;
    async_copy_b128(&vs[buf][r * LDB + c], &vg[(size_t)r * SEQ + t0k + c]);
  };

  v8f acc[2][2] = {};

  auto compute = [&](const unsigned short* psb, const unsigned short* vsb) {
    Frag16 a[2], bf[2];
    int am = wm + (lane & 15);
    int bn = wn + (lane & 15);
#pragma unroll
    for (int ti = 0; ti < 2; ++ti) {
      load_a_frag(a[ti], psb, am + ti * 16, lane);
      load_b_frag(bf[ti], vsb, bn + ti * 16, lane);
    }
#pragma unroll
    for (int ti = 0; ti < 2; ++ti)
#pragma unroll
      for (int tj = 0; tj < 2; ++tj)
        acc[ti][tj] = __builtin_amdgcn_wmma_f32_16x16x32_bf16(
            false, a[ti].v, false, bf[tj].v, (short)0, acc[ti][tj],
            false, false);
  };

  const int NIT = SEQ / TK;                      // 64 (even)
  stage(0, 0);
  for (int it = 0; it < NIT - 2; it += 2) {
    wait_async0(); __syncthreads();
    stage(1, (it + 1) * TK);
    compute(ps[0], vs[0]);
    wait_async0(); __syncthreads();
    stage(0, (it + 2) * TK);
    compute(ps[1], vs[1]);
  }
  wait_async0(); __syncthreads();
  stage(1, (NIT - 1) * TK);
  compute(ps[0], vs[0]);
  wait_async0(); __syncthreads();
  compute(ps[1], vs[1]);

#pragma unroll
  for (int ti = 0; ti < 2; ++ti)
#pragma unroll
    for (int tj = 0; tj < 2; ++tj) {
      int h = h0 + wn + tj * 16 + (lane & 15);
#pragma unroll
      for (int r = 0; r < 8; ++r) {
        int s = s0 + wm + ti * 16 + ((lane < 16) ? 0 : 8) + r;
        O[((size_t)b * SEQ + s) * HID + h] = acc[ti][tj][r];
      }
    }
}

// ---------------- launch ----------------
extern "C" void kernel_launch(void* const* d_in, const int* in_sizes, int n_in,
                              void* d_out, int out_size, void* d_ws,
                              size_t ws_size, hipStream_t stream) {
  (void)in_sizes; (void)n_in; (void)out_size; (void)ws_size;
  const float* x    = (const float*)d_in[0];
  const int*   mask = (const int*)d_in[1];
  const float* wq   = (const float*)d_in[2];
  const float* bq   = (const float*)d_in[3];
  const float* wk   = (const float*)d_in[4];
  const float* bk   = (const float*)d_in[5];
  const float* wv   = (const float*)d_in[6];
  const float* bv   = (const float*)d_in[7];
  float* out = (float*)d_out;

  unsigned short* xb  = (unsigned short*)d_ws;                 // MTOT*HID
  unsigned short* wqT = xb  + (size_t)MTOT * HID;              // HID*HID
  unsigned short* wkT = wqT + (size_t)HID * HID;
  unsigned short* wvT = wkT + (size_t)HID * HID;
  unsigned short* Qb  = wvT + (size_t)HID * HID;               // MTOT*HID
  unsigned short* Kb  = Qb  + (size_t)MTOT * HID;
  unsigned short* Vt  = Kb  + (size_t)MTOT * HID;              // [B][H][S]
  unsigned short* P   = Vt  + (size_t)MTOT * HID;              // B*S*S

  int nx = MTOT * HID;
  k_convert<<<nx / 4 / 256, 256, 0, stream>>>(x, xb, nx);

  dim3 tb(32, 8), tg(HID / 32, HID / 32);
  k_transpose_convert<<<tg, tb, 0, stream>>>(wq, wqT);
  k_transpose_convert<<<tg, tb, 0, stream>>>(wk, wkT);
  k_transpose_convert<<<tg, tb, 0, stream>>>(wv, wvT);

  k_qk_gemm<<<dim3(MTOT / TM, HID / TN), 256, 0, stream>>>(
      xb, wqT, wkT, bq, bk, Qb, Kb);
  k_v_gemm<<<dim3(MTOT / TM, HID / TN), 256, 0, stream>>>(xb, wvT, bv, Vt);

  k_scores<<<dim3(SEQ / TM, SEQ / TN, BATCH), 256, 0, stream>>>(Qb, Kb, mask, P);

  k_softmax<<<BATCH * SEQ, 256, 0, stream>>>(P);

  k_out_gemm<<<dim3(SEQ / TM, HID / TN, BATCH), 256, 0, stream>>>(P, Vt, out);
}
